// Mamba1Mixer_25503515803770
// MI455X (gfx1250) — matrix operand order repaired and checked by reading, not compiled, not run on hardware
//
#include <hip/hip_runtime.h>
#include <cstdint>
#include <cstddef>

// ---------------------------------------------------------------------------
// Problem constants (match reference)
// ---------------------------------------------------------------------------
#define BB      2
#define LL      2048
#define DD      1024
#define DI      2048
#define DTRANK  64
#define NSTATE  16
#define DCONV   4
#define MROWS   (BB * LL)        // 4096 token rows

typedef __attribute__((ext_vector_type(16))) __bf16 v16bf;
typedef __attribute__((ext_vector_type(8)))  float  v8f;
typedef __attribute__((ext_vector_type(4)))  unsigned int v4u;

union FragU {
    v4u   u4[2];
    v16bf bf;
};

static __device__ __forceinline__ unsigned short f2bf(float f) {
    unsigned int u = __builtin_bit_cast(unsigned int, f);
    unsigned int r = u + 0x7FFFu + ((u >> 16) & 1u);   // round-to-nearest-even
    return (unsigned short)(r >> 16);
}

static __device__ __forceinline__ float silu_f(float x) {
    return x / (1.0f + __expf(-x));
}

// ---------------------------------------------------------------------------
// f32 -> bf16 conversion kernels
// ---------------------------------------------------------------------------
__global__ void cvt_f32_bf16_kernel(const float* __restrict__ src,
                                    unsigned short* __restrict__ dst, int n) {
    int i = blockIdx.x * blockDim.x + threadIdx.x;
    if (i < n) dst[i] = f2bf(src[i]);
}

// strided slice: dst[r*cols + c] = bf16(src[r*srcLd + srcOff + c])
__global__ void cvt_slice_bf16_kernel(const float* __restrict__ src,
                                      unsigned short* __restrict__ dst,
                                      int rows, int cols, int srcLd, int srcOff) {
    int i = blockIdx.x * blockDim.x + threadIdx.x;
    if (i >= rows * cols) return;
    int r = i / cols, c = i - r * cols;
    dst[i] = f2bf(src[(size_t)r * srcLd + srcOff + c]);
}

// ---------------------------------------------------------------------------
// WMMA bf16 NT GEMM: C[M,N](f32) = A[M,K](bf16) * B[N,K](bf16)^T
// Register-blocked: each wave owns a (BM*16) x (BN*16) macro-tile and issues
// BM*BN v_wmma_f32_16x16x32_bf16 per K-step of 32, reusing each A fragment
// across BN tiles and each B fragment across BM tiles.
// N, K compile-time so fragment addresses fold into instruction offsets.
// ---------------------------------------------------------------------------
template <int N, int K, int BM, int BN>
__global__ __launch_bounds__(256) void
gemm_nt_bf16_kernel(const unsigned short* __restrict__ A,
                    const unsigned short* __restrict__ Bmat,
                    float* __restrict__ C, int M) {
    const int lane   = threadIdx.x & 31;
    const int wave   = threadIdx.x >> 5;
    const int tilesN = N / (16 * BN);
    const int nTiles = (M / (16 * BM)) * tilesN;
    const int tile   = blockIdx.x * (blockDim.x >> 5) + wave;
    if (tile >= nTiles) return;                       // wave-uniform branch
    const int sM   = tile / tilesN;
    const int sN   = tile - sM * tilesN;
    const int half = lane >> 4;                       // 0 or 1
    const int mn   = lane & 15;

    // Per-lane base rows (frag row i/j adds i*16*K elements, a compile-time
    // multiple -> folds into the global_load_b128 immediate offset).
    const unsigned short* aBase = A    + (size_t)(sM * BM * 16 + mn) * K;
    const unsigned short* bBase = Bmat + (size_t)(sN * BN * 16 + mn) * K;
    const int aoff = half * 8;    // A frag: elems 0-7 at K=k0+aoff, 8-15 at +16
    const int boff = half * 8;    // B frag: elems 0-7 at K=k0+boff, 8-15 at +16

    v8f acc[BM][BN];
#pragma unroll
    for (int i = 0; i < BM; ++i)
#pragma unroll
        for (int j = 0; j < BN; ++j)
            acc[i][j] = (v8f){0.f, 0.f, 0.f, 0.f, 0.f, 0.f, 0.f, 0.f};

    for (int k0 = 0; k0 < K; k0 += 32) {
        FragU fa[BM], fb[BN];
#pragma unroll
        for (int i = 0; i < BM; ++i) {
            const unsigned short* p = aBase + (size_t)i * 16 * K + k0 + aoff;
            fa[i].u4[0] = *(const v4u*)(p);
            fa[i].u4[1] = *(const v4u*)(p + 16);
        }
#pragma unroll
        for (int j = 0; j < BN; ++j) {
            const unsigned short* p = bBase + (size_t)j * 16 * K + k0 + boff;
            fb[j].u4[0] = *(const v4u*)(p);
            fb[j].u4[1] = *(const v4u*)(p + 16);
        }
#pragma unroll
        for (int i = 0; i < BM; ++i)
#pragma unroll
            for (int j = 0; j < BN; ++j)
                acc[i][j] = __builtin_amdgcn_wmma_f32_16x16x32_bf16(
                    /*neg_a=*/false, fa[i].bf, /*neg_b=*/false, fb[j].bf,
                    /*c_mod=*/(short)0, acc[i][j],
                    /*reuse_a=*/false, /*reuse_b=*/false);
    }

#pragma unroll
    for (int i = 0; i < BM; ++i) {
#pragma unroll
        for (int j = 0; j < BN; ++j) {
            float* cBase = C + (size_t)(sM * BM * 16 + i * 16 + half * 8) * N
                             + sN * BN * 16 + j * 16 + mn;
#pragma unroll
            for (int r = 0; r < 8; ++r) cBase[(size_t)r * N] = acc[i][j][r];
        }
    }
}

// ---------------------------------------------------------------------------
// Causal depthwise conv (DCONV=4) + bias + silu over xi half of xz.
// xz: (MROWS, 2*DI) row-major; writes xi f32 and xi bf16, both (MROWS, DI).
// ---------------------------------------------------------------------------
__global__ void conv_silu_kernel(const float* __restrict__ xz,
                                 const float* __restrict__ cw,
                                 const float* __restrict__ cb,
                                 float* __restrict__ xi,
                                 unsigned short* __restrict__ xi_bf) {
    int idx = blockIdx.x * blockDim.x + threadIdx.x;   // over MROWS*DI
    if (idx >= MROWS * DI) return;
    int d   = idx & (DI - 1);
    int row = idx >> 11;                               // DI == 2048
    int l   = row & (LL - 1);                          // LL == 2048
    float acc = cb[d];
#pragma unroll
    for (int j = 0; j < DCONV; ++j) {
        int lp = l - (DCONV - 1) + j;
        if (lp >= 0)
            acc += cw[d * DCONV + j] * xz[(size_t)(row - (DCONV - 1) + j) * (2 * DI) + d];
    }
    float s = silu_f(acc);
    xi[idx]    = s;
    xi_bf[idx] = f2bf(s);
}

// ---------------------------------------------------------------------------
// dt = softplus(dt_raw + b_dt), in place on (MROWS, DI)
// ---------------------------------------------------------------------------
__global__ void softplus_bias_kernel(float* __restrict__ dt,
                                     const float* __restrict__ b_dt) {
    int idx = blockIdx.x * blockDim.x + threadIdx.x;
    if (idx >= MROWS * DI) return;
    int d = idx & (DI - 1);
    float v = dt[idx] + b_dt[d];
    dt[idx] = (v > 20.0f) ? v : log1pf(__expf(v));
}

// ---------------------------------------------------------------------------
// Selective scan. Each 16-lane half-wave owns one channel (b,d); lane = state n.
// h_n <- h_n * exp(dt*A_n) + dt*B_n*xi ; y = sum_n h_n*C_n + D_skip*xi
// Reduction over n via xor-shuffles (masks 1,2,4,8 stay inside the half-wave).
// Software-pipelined: next step's loads are issued before this step's math so
// global latency overlaps the v_exp + shuffle chain.
// ---------------------------------------------------------------------------
__global__ void selective_scan_kernel(const float* __restrict__ xi,
                                      const float* __restrict__ dt,
                                      const float* __restrict__ xdbl,  // (MROWS,96)
                                      const float* __restrict__ A_log, // (DI,16)
                                      const float* __restrict__ D_skip,
                                      float* __restrict__ y) {
    const int lane = threadIdx.x & 31;
    const int wave = threadIdx.x >> 5;
    const int half = lane >> 4;
    const int n    = lane & 15;
    const int chan = (blockIdx.x * (blockDim.x >> 5) + wave) * 2 + half;
    if (chan >= BB * DI) return;
    const int b = chan >> 11;            // DI == 2048
    const int d = chan & (DI - 1);

    const float An = -__expf(A_log[d * NSTATE + n]);
    const float Dk = D_skip[d];
    const int   XL = DTRANK + 2 * NSTATE;            // 96

    const size_t row0 = (size_t)b * LL;
    float dt_v = dt[row0 * DI + d];
    float xi_v = xi[row0 * DI + d];
    float Bn   = xdbl[row0 * XL + DTRANK + n];
    float Cn   = xdbl[row0 * XL + DTRANK + NSTATE + n];

    float h = 0.0f;
    for (int l = 0; l < LL; ++l) {
        const size_t row = row0 + l;
        float dt_nx = 0.f, xi_nx = 0.f, Bn_nx = 0.f, Cn_nx = 0.f;
        if (l + 1 < LL) {                            // prefetch next step
            dt_nx = dt[(row + 1) * DI + d];
            xi_nx = xi[(row + 1) * DI + d];
            Bn_nx = xdbl[(row + 1) * XL + DTRANK + n];
            Cn_nx = xdbl[(row + 1) * XL + DTRANK + NSTATE + n];
        }
        const float dA = __expf(dt_v * An);
        h = h * dA + dt_v * Bn * xi_v;
        float p = h * Cn;
        p += __shfl_xor(p, 1, 32);
        p += __shfl_xor(p, 2, 32);
        p += __shfl_xor(p, 4, 32);
        p += __shfl_xor(p, 8, 32);
        if (n == 0) y[row * DI + d] = p + Dk * xi_v;
        dt_v = dt_nx; xi_v = xi_nx; Bn = Bn_nx; Cn = Cn_nx;
    }
}

// ---------------------------------------------------------------------------
// Gate: yz_bf = bf16( y * silu(z) ), z = xz[:, DI:2*DI]
// ---------------------------------------------------------------------------
__global__ void gate_kernel(const float* __restrict__ y,
                            const float* __restrict__ xz,
                            unsigned short* __restrict__ yz_bf) {
    int idx = blockIdx.x * blockDim.x + threadIdx.x;
    if (idx >= MROWS * DI) return;
    int d   = idx & (DI - 1);
    int row = idx >> 11;
    float z = xz[(size_t)row * (2 * DI) + DI + d];
    yz_bf[idx] = f2bf(y[idx] * silu_f(z));
}

// ---------------------------------------------------------------------------
// Host side
// ---------------------------------------------------------------------------
extern "C" void kernel_launch(void* const* d_in, const int* in_sizes, int n_in,
                              void* d_out, int out_size, void* d_ws, size_t ws_size,
                              hipStream_t stream) {
    (void)in_sizes; (void)n_in; (void)out_size; (void)ws_size;

    const float* x      = (const float*)d_in[0];   // (B,L,D)
    const float* W_in   = (const float*)d_in[1];   // (2*DI, D)
    const float* conv_w = (const float*)d_in[2];   // (DI,1,4)
    const float* conv_b = (const float*)d_in[3];   // (DI,)
    const float* W_x    = (const float*)d_in[4];   // (96, DI)
    const float* W_dt   = (const float*)d_in[5];   // (DI, 64)
    const float* b_dt   = (const float*)d_in[6];   // (DI,)
    const float* W_out  = (const float*)d_in[7];   // (D, DI)
    const float* A_log  = (const float*)d_in[8];   // (DI, 16)
    const float* D_skip = (const float*)d_in[9];   // (DI,)
    float* out = (float*)d_out;                    // (B,L,D)

    // bump allocator over workspace
    char* ws = (char*)d_ws;
    size_t off = 0;
    auto alloc = [&](size_t bytes) -> void* {
        void* p = ws + off;
        off = (off + bytes + 255) & ~(size_t)255;
        return p;
    };

    unsigned short* x_bf     = (unsigned short*)alloc((size_t)MROWS * DD * 2);
    unsigned short* Win_bf   = (unsigned short*)alloc((size_t)2 * DI * DD * 2);
    unsigned short* Wx_bf    = (unsigned short*)alloc((size_t)(DTRANK + 2 * NSTATE) * DI * 2);
    unsigned short* Wdt_bf   = (unsigned short*)alloc((size_t)DI * DTRANK * 2);
    unsigned short* Wout_bf  = (unsigned short*)alloc((size_t)DD * DI * 2);
    float*          xz       = (float*)alloc((size_t)MROWS * 2 * DI * 4);
    float*          xi       = (float*)alloc((size_t)MROWS * DI * 4);
    unsigned short* xi_bf    = (unsigned short*)alloc((size_t)MROWS * DI * 2);
    float*          x_dbl    = (float*)alloc((size_t)MROWS * (DTRANK + 2 * NSTATE) * 4);
    unsigned short* dtin_bf  = (unsigned short*)alloc((size_t)MROWS * DTRANK * 2);
    float*          dt       = (float*)alloc((size_t)MROWS * DI * 4);
    float*          y        = (float*)alloc((size_t)MROWS * DI * 4);
    unsigned short* yz_bf    = (unsigned short*)alloc((size_t)MROWS * DI * 2);

    const int T = 256;
    auto cdiv = [](int a, int b) { return (a + b - 1) / b; };

    // 1. convert inputs to bf16
    cvt_f32_bf16_kernel<<<cdiv(MROWS * DD, T), T, 0, stream>>>(x, x_bf, MROWS * DD);
    cvt_f32_bf16_kernel<<<cdiv(2 * DI * DD, T), T, 0, stream>>>(W_in, Win_bf, 2 * DI * DD);
    cvt_f32_bf16_kernel<<<cdiv((DTRANK + 2 * NSTATE) * DI, T), T, 0, stream>>>(
        W_x, Wx_bf, (DTRANK + 2 * NSTATE) * DI);
    cvt_f32_bf16_kernel<<<cdiv(DI * DTRANK, T), T, 0, stream>>>(W_dt, Wdt_bf, DI * DTRANK);
    cvt_f32_bf16_kernel<<<cdiv(DD * DI, T), T, 0, stream>>>(W_out, Wout_bf, DD * DI);

    // 2. xz = x @ W_in^T   (4096 x 4096, K=1024), 64x64 per wave
    {
        int tiles = (MROWS / 64) * ((2 * DI) / 64);           // 4096
        gemm_nt_bf16_kernel<2 * DI, DD, 4, 4>
            <<<cdiv(tiles, 8), T, 0, stream>>>(x_bf, Win_bf, xz, MROWS);
    }

    // 3. causal depthwise conv + silu -> xi (f32 + bf16)
    conv_silu_kernel<<<cdiv(MROWS * DI, T), T, 0, stream>>>(xz, conv_w, conv_b, xi, xi_bf);

    // 4. x_dbl = xi @ W_x^T   (4096 x 96, K=2048), fine-grained for parallelism
    {
        int tiles = (MROWS / 16) * ((DTRANK + 2 * NSTATE) / 16);  // 1536
        gemm_nt_bf16_kernel<DTRANK + 2 * NSTATE, DI, 1, 1>
            <<<cdiv(tiles, 8), T, 0, stream>>>(xi_bf, Wx_bf, x_dbl, MROWS);
    }

    // 5. dt_in = bf16(x_dbl[:, :64])
    cvt_slice_bf16_kernel<<<cdiv(MROWS * DTRANK, T), T, 0, stream>>>(
        x_dbl, dtin_bf, MROWS, DTRANK, DTRANK + 2 * NSTATE, 0);

    // 6. dt_raw = dt_in @ W_dt^T   (4096 x 2048, K=64), 64x64 per wave
    {
        int tiles = (MROWS / 64) * (DI / 64);                 // 2048
        gemm_nt_bf16_kernel<DI, DTRANK, 4, 4>
            <<<cdiv(tiles, 8), T, 0, stream>>>(dtin_bf, Wdt_bf, dt, MROWS);
    }

    // 7. dt = softplus(dt_raw + b_dt)
    softplus_bias_kernel<<<cdiv(MROWS * DI, T), T, 0, stream>>>(dt, b_dt);

    // 8. selective scan -> y
    {
        int chans = BB * DI;                 // 4096; 2 per wave, 8 waves per block
        int blocks = chans / 16;             // 256
        selective_scan_kernel<<<blocks, T, 0, stream>>>(xi, dt, x_dbl, A_log, D_skip, y);
    }

    // 9. yz = bf16( y * silu(z) )
    gate_kernel<<<cdiv(MROWS * DI, T), T, 0, stream>>>(y, xz, yz_bf);

    // 10. out = yz @ W_out^T   (4096 x 1024, K=2048), 64x64 per wave
    {
        int tiles = (MROWS / 64) * (DD / 64);                 // 1024
        gemm_nt_bf16_kernel<DD, DI, 4, 4>
            <<<cdiv(tiles, 8), T, 0, stream>>>(yz_bf, Wout_bf, out, MROWS);
    }
}
